// QNN_14431090114798
// MI455X (gfx1250) — matrix operand order repaired and checked by reading, not compiled, not run on hardware
//
#include <hip/hip_runtime.h>

// ---------------------------------------------------------------------------
// QNN = fixed 64x64 complex unitary U (from weights) applied per batch row,
// then |.|^2, then a 64x4 contraction.  out = (|U x|^2) @ M / ||x||^2 + b.
// Main GEMM: [262144 x 64] x [64 x 128] bf16 WMMA (f32 accum) -> memory bound
// at 23.3 TB/s (~64MB of X).  Epilogue GEMM: probs[16x64] x Mpad[64x16] WMMA.
// X loads are software-pipelined one 16-row tile ahead.
// ---------------------------------------------------------------------------

typedef __attribute__((ext_vector_type(16))) __bf16          v16bf;
typedef __attribute__((ext_vector_type(8)))  __bf16          v8bf;
typedef __attribute__((ext_vector_type(8)))  float           v8f;
typedef __attribute__((ext_vector_type(4)))  int             v4i;

#if defined(__has_builtin)
#if __has_builtin(__builtin_amdgcn_global_load_async_to_lds_b128)
#define HAVE_ASYNC_LDS 1
#endif
#if __has_builtin(__builtin_amdgcn_s_wait_asynccnt)
#define HAVE_WAIT_ASYNC 1
#endif
#endif

__device__ __forceinline__ unsigned short f2bf_bits(float f) {
  unsigned int x = __float_as_uint(f);
  x += 0x7FFFu + ((x >> 16) & 1u);   // round-to-nearest-even
  return (unsigned short)(x >> 16);
}

// ---------------------------------------------------------------------------
// Setup: build U (simulate circuit on 64 basis columns), emit bf16 WMMA
// B-fragments for [K=64 j][N=128 (re|im of U^T)] plus Mpad[K=64][N=16].
// Fragment layout (ISA 16-bit B 32x16): block=(tileN*2+kstep), within block:
// lane = (N%16) + 16*(((K%32)/8)%2), halfword = (K%8) + 8*((K%32)/16).
// ---------------------------------------------------------------------------
__global__ __launch_bounds__(64) void qnn_build_frags(
    const float* __restrict__ w, const float* __restrict__ fcw,
    unsigned short* __restrict__ frag) {
  __shared__ float Sr[64 * 64];
  __shared__ float Si[64 * 64];
  const int j = threadIdx.x;                 // this thread owns column j
  for (int i = 0; i < 64; ++i) { Sr[i*64+j] = (i == j) ? 1.f : 0.f; Si[i*64+j] = 0.f; }

  for (int l = 0; l < 8; ++l) {
    for (int q = 0; q < 6; ++q) {
      const float phi = w[(l*6+q)*3+0];
      const float th  = w[(l*6+q)*3+1];
      const float om  = w[(l*6+q)*3+2];
      float sh, ch; __sincosf(0.5f*th,       &sh, &ch);
      float sa, ca; __sincosf(0.5f*(phi+om), &sa, &ca);
      float sb, cb; __sincosf(0.5f*(phi-om), &sb, &cb);
      const float u00r =  ca*ch, u00i = -sa*ch;
      const float u01r = -cb*sh, u01i = -sb*sh;
      const float u10r =  cb*sh, u10i = -sb*sh;
      const float u11r =  ca*ch, u11i =  sa*ch;
      const int mask = 1 << (5 - q);          // qubit 0 = MSB of basis index
      for (int i0 = 0; i0 < 64; ++i0) {
        if (i0 & mask) continue;
        const int i1 = i0 | mask;
        const float o0r = Sr[i0*64+j], o0i = Si[i0*64+j];
        const float o1r = Sr[i1*64+j], o1i = Si[i1*64+j];
        Sr[i0*64+j] = u00r*o0r - u00i*o0i + u01r*o1r - u01i*o1i;
        Si[i0*64+j] = u00r*o0i + u00i*o0r + u01r*o1i + u01i*o1r;
        Sr[i1*64+j] = u10r*o0r - u10i*o0i + u11r*o1r - u11i*o1i;
        Si[i1*64+j] = u10r*o0i + u10i*o0r + u11r*o1i + u11i*o1r;
      }
    }
    const int rr = (l % 5) + 1;               // entangler range
    for (int q = 0; q < 6; ++q) {
      const int mc = 1 << (5 - q);
      const int mt = 1 << (5 - ((q + rr) % 6));
      for (int i = 0; i < 64; ++i) {
        if ((i & mc) && !(i & mt)) {          // control=1, target=0 -> swap pair
          const int i2 = i | mt;
          float t0 = Sr[i*64+j]; Sr[i*64+j] = Sr[i2*64+j]; Sr[i2*64+j] = t0;
          float t1 = Si[i*64+j]; Si[i*64+j] = Si[i2*64+j]; Si[i2*64+j] = t1;
        }
      }
    }
  }

  // B fragments: B[K=j][N=n] = (n<64 ? Re U[n][j] : Im U[n-64][j])
  for (int n = 0; n < 128; ++n) {
    const float v = (n < 64) ? Sr[n*64+j] : Si[(n-64)*64+j];
    const int t  = n >> 4;
    const int s  = j >> 5;
    const int jj = j & 31;
    const int off = (t*2 + s)*512 + ((n & 15) + (((jj >> 3) & 1) << 4))*16 + (jj & 7) + ((jj >> 4) << 3);
    frag[off] = f2bf_bits(v);
  }
  // Mpad fragments: M[k][c] = sum_q Z(k,q)*fc_w[c][q], c<4 else 0
  {
    const int k = j, s = k >> 5, kk = k & 31;
    for (int c = 0; c < 16; ++c) {
      float v = 0.f;
      if (c < 4)
        for (int q = 0; q < 6; ++q) {
          const int bit = (k >> (5 - q)) & 1;
          v += (bit ? -1.f : 1.f) * fcw[c*6+q];
        }
      frag[8192 + s*512 + (c + (((kk >> 3) & 1) << 4))*16 + (kk & 7) + ((kk >> 4) << 3)] = f2bf_bits(v);
    }
  }
}

// ---------------------------------------------------------------------------
// Main kernel: each wave (wave32) processes 16-row tiles of X.
// ---------------------------------------------------------------------------
__global__ __launch_bounds__(256) void qnn_main(
    const float* __restrict__ X, const unsigned short* __restrict__ frag,
    const float* __restrict__ fcb, float* __restrict__ out, int ntiles) {
  __shared__ __align__(32) __bf16 ldsB[9216];       // 16KB B + 2KB Mpad frags
  __shared__ __align__(32) __bf16 probs[8 * 1024];  // per-wave 16x64 bf16

  const int tid = threadIdx.x;
  // Stage all fragments to LDS (async-to-LDS path when toolchain exposes it).
#if defined(HAVE_ASYNC_LDS)
  {
    const char* src = (const char*)frag;
    for (int i = tid; i < 1152; i += 256)
      __builtin_amdgcn_global_load_async_to_lds_b128(
          (v4i*)(src + (size_t)i * 16), (v4i*)&ldsB[i * 8], 0, 0);
#if defined(HAVE_WAIT_ASYNC)
    __builtin_amdgcn_s_wait_asynccnt(0);
#else
    asm volatile("s_wait_asynccnt 0" ::: "memory");
#endif
  }
#else
  {
    const uint4* src = (const uint4*)frag;
    uint4* dst = (uint4*)ldsB;
    for (int i = tid; i < 1152; i += 256) dst[i] = src[i];
  }
#endif
  __syncthreads();

  const int wid  = tid >> 5;
  const int L    = tid & 31;
  const int m    = L & 15;     // row within tile / output column
  const int half = L >> 4;

  const v16bf mh0 = *(const v16bf*)&ldsB[8192 +       L*16];  // Mpad kstep 0
  const v16bf mh1 = *(const v16bf*)&ldsB[8192 + 512 + L*16];  // Mpad kstep 1
  const float bias = (m < 4) ? fcb[m] : 0.f;
  __bf16* myprobs = &probs[wid * 1024];
  const v8f zero = {0.f, 0.f, 0.f, 0.f, 0.f, 0.f, 0.f, 0.f};
  const int stride = (int)gridDim.x * 8;
  const int tile0  = blockIdx.x*8 + wid;

  // ---- software pipeline: cur[] holds this tile's X rows in registers ----
  float4 cur[8];
  {
    const int t0 = (tile0 < ntiles) ? tile0 : (ntiles - 1);
    const float* xr = X + (size_t)(t0*16 + m) * 64;
    #pragma unroll
    for (int s = 0; s < 2; ++s) {
      const int c0 = s*32 + half*8;
      cur[s*4+0] = *(const float4*)(xr + c0);
      cur[s*4+1] = *(const float4*)(xr + c0 + 4);
      cur[s*4+2] = *(const float4*)(xr + c0 + 16);
      cur[s*4+3] = *(const float4*)(xr + c0 + 20);
    }
  }

  for (int tile = tile0; tile < ntiles; tile += stride) {
    // Issue next tile's loads first; clamp address so the loads are
    // unconditional (no divergence, always a valid row).
    const int nt = tile + stride;
    const int ntc = (nt < ntiles) ? nt : (ntiles - 1);
    const float* xn = X + (size_t)(ntc*16 + m) * 64;
    float4 nxt[8];
    #pragma unroll
    for (int s = 0; s < 2; ++s) {
      const int c0 = s*32 + half*8;
      nxt[s*4+0] = *(const float4*)(xn + c0);
      nxt[s*4+1] = *(const float4*)(xn + c0 + 4);
      nxt[s*4+2] = *(const float4*)(xn + c0 + 16);
      nxt[s*4+3] = *(const float4*)(xn + c0 + 20);
    }
    // Warm GL2 two strides out (next stride is covered by the pipeline).
    if (tile + 2*stride < ntiles)
      __builtin_prefetch(X + (size_t)((tile + 2*stride)*16 + m) * 64, 0, 1);

    // ---- convert this tile (already in registers) to A fragments ----
    // Two independent even/odd FMA chains -> packs into v_pk_fma_f32.
    float n0 = 0.f, n1 = 0.f;
    v16bf A[2];
    #pragma unroll
    for (int s = 0; s < 2; ++s) {            // K-step: A covers K = s*32..s*32+31
      const float4 a0 = cur[s*4+0], a1 = cur[s*4+1];
      const float4 b0 = cur[s*4+2], b1 = cur[s*4+3];
      n0 = fmaf(a0.x, a0.x, n0); n1 = fmaf(a0.y, a0.y, n1);
      n0 = fmaf(a0.z, a0.z, n0); n1 = fmaf(a0.w, a0.w, n1);
      n0 = fmaf(a1.x, a1.x, n0); n1 = fmaf(a1.y, a1.y, n1);
      n0 = fmaf(a1.z, a1.z, n0); n1 = fmaf(a1.w, a1.w, n1);
      n0 = fmaf(b0.x, b0.x, n0); n1 = fmaf(b0.y, b0.y, n1);
      n0 = fmaf(b0.z, b0.z, n0); n1 = fmaf(b0.w, b0.w, n1);
      n0 = fmaf(b1.x, b1.x, n0); n1 = fmaf(b1.y, b1.y, n1);
      n0 = fmaf(b1.z, b1.z, n0); n1 = fmaf(b1.w, b1.w, n1);
      v16bf t;
      t[0]=(__bf16)a0.x; t[1]=(__bf16)a0.y; t[2]=(__bf16)a0.z; t[3]=(__bf16)a0.w;
      t[4]=(__bf16)a1.x; t[5]=(__bf16)a1.y; t[6]=(__bf16)a1.z; t[7]=(__bf16)a1.w;
      t[8]=(__bf16)b0.x; t[9]=(__bf16)b0.y; t[10]=(__bf16)b0.z; t[11]=(__bf16)b0.w;
      t[12]=(__bf16)b1.x; t[13]=(__bf16)b1.y; t[14]=(__bf16)b1.z; t[15]=(__bf16)b1.w;
      A[s] = t;
    }
    float nrm = n0 + n1;
    // ||x||^2: lane L holds half the row; partner is L^16 (wave32 shuffle)
    nrm += __shfl_xor(nrm, 16, 32);
    const float rv = __builtin_amdgcn_rcpf(fmaxf(nrm, 1e-24f));
    float rs[8];                              // 1/||x||^2 for row r+8*half
    #pragma unroll
    for (int r = 0; r < 8; ++r) rs[r] = __shfl(rv, r + 8*half, 32);

    #pragma unroll
    for (int p = 0; p < 4; ++p) {            // N-tile pair p (re) / p+4 (im)
      const v16bf br0 = *(const v16bf*)&ldsB[((p)*2 + 0)*512 + L*16];
      const v16bf br1 = *(const v16bf*)&ldsB[((p)*2 + 1)*512 + L*16];
      const v16bf bi0 = *(const v16bf*)&ldsB[((p+4)*2 + 0)*512 + L*16];
      const v16bf bi1 = *(const v16bf*)&ldsB[((p+4)*2 + 1)*512 + L*16];
      v8f cr = __builtin_amdgcn_wmma_f32_16x16x32_bf16(false, A[0], false, br0, (short)0, zero, false, false);
      cr     = __builtin_amdgcn_wmma_f32_16x16x32_bf16(false, A[1], false, br1, (short)0, cr,   false, false);
      v8f ci = __builtin_amdgcn_wmma_f32_16x16x32_bf16(false, A[0], false, bi0, (short)0, zero, false, false);
      ci     = __builtin_amdgcn_wmma_f32_16x16x32_bf16(false, A[1], false, bi1, (short)0, ci,   false, false);
      #pragma unroll
      for (int r = 0; r < 8; ++r) {
        // unnormalized |y|^2 -- the 1/||x||^2 row scale commutes to the end
        myprobs[(r + 8*half)*64 + p*16 + m] = (__bf16)(cr[r]*cr[r] + ci[r]*ci[r]);
      }
    }

    // second GEMM: probs[16x64] x Mpad[64x16]  (same-wave LDS, no barrier)
    v16bf A2[2];
    #pragma unroll
    for (int s = 0; s < 2; ++s) {
      const int c0 = s*32 + half*8;
      const v8bf lo = *(const v8bf*)&myprobs[m*64 + c0];
      const v8bf hi = *(const v8bf*)&myprobs[m*64 + c0 + 16];
      A2[s] = __builtin_shufflevector(lo, hi, 0,1,2,3,4,5,6,7,8,9,10,11,12,13,14,15);
    }
    v8f c2 = __builtin_amdgcn_wmma_f32_16x16x32_bf16(false, A2[0], false, mh0, (short)0, zero, false, false);
    c2     = __builtin_amdgcn_wmma_f32_16x16x32_bf16(false, A2[1], false, mh1, (short)0, c2,   false, false);

    if (m < 4) {
      #pragma unroll
      for (int r = 0; r < 8; ++r)
        out[(size_t)(tile*16 + r + 8*half)*4 + m] = fmaf(c2[r], rs[r], bias);
    }

    #pragma unroll
    for (int i = 0; i < 8; ++i) cur[i] = nxt[i];   // rotate pipeline
  }
}

extern "C" void kernel_launch(void* const* d_in, const int* in_sizes, int n_in,
                              void* d_out, int out_size, void* d_ws, size_t ws_size,
                              hipStream_t stream) {
  const float* x   = (const float*)d_in[0];
  const float* w   = (const float*)d_in[1];
  const float* fcw = (const float*)d_in[2];
  const float* fcb = (const float*)d_in[3];
  float* out = (float*)d_out;
  unsigned short* frag = (unsigned short*)d_ws;   // 18KB of bf16 fragments

  qnn_build_frags<<<1, 64, 0, stream>>>(w, fcw, frag);

  const int batch  = in_sizes[0] / 64;
  const int ntiles = batch / 16;
  qnn_main<<<512, 256, 0, stream>>>(x, frag, fcb, out, ntiles);
}
